// InferenceNet_35218731828124
// MI455X (gfx1250) — matrix operand order repaired and checked by reading, not compiled, not run on hardware
//
#include <hip/hip_runtime.h>
#include <hip/hip_bf16.h>

// ---------------- problem constants ----------------
#define B_    4096
#define T_    20
#define N_IN  38
#define ENC   128
#define HID   256
#define ZD    64
#define CD    16
#define BT    (B_ * T_)

typedef __attribute__((ext_vector_type(16))) _Float16 v16h;
typedef __attribute__((ext_vector_type(8)))  float    v8f;

__device__ __forceinline__ float lrelu(float x) { return x >= 0.f ? x : 0.2f * x; }
__device__ __forceinline__ float sigm(float x)  { return 1.f / (1.f + __expf(-x)); }

// =====================================================================
// Encoder conv front-end: per-sample conv1d chain (1->8->16->32, k=2,s=2)
// weights staged in LDS; emits f16 features [BT,160] for the WMMA GEMMs.
// =====================================================================
__global__ __launch_bounds__(256) void conv_enc_kernel(
    const float* __restrict__ x,
    const float* __restrict__ c0w, const float* __restrict__ c0b,
    const float* __restrict__ c1w, const float* __restrict__ c1b,
    const float* __restrict__ c2w, const float* __restrict__ c2b,
    _Float16* __restrict__ feat)
{
    __shared__ float smem[16 + 8 + 256 + 16 + 1024 + 32];
    float* s0w = smem;
    float* s0b = s0w + 16;
    float* s1w = s0b + 8;
    float* s1b = s1w + 256;
    float* s2w = s1b + 16;
    float* s2b = s2w + 1024;
    int tid = threadIdx.x;
    for (int i = tid; i < 16;   i += 256) s0w[i] = c0w[i];
    for (int i = tid; i < 8;    i += 256) s0b[i] = c0b[i];
    for (int i = tid; i < 256;  i += 256) s1w[i] = c1w[i];
    for (int i = tid; i < 16;   i += 256) s1b[i] = c1b[i];
    for (int i = tid; i < 1024; i += 256) s2w[i] = c2w[i];
    for (int i = tid; i < 32;   i += 256) s2b[i] = c2b[i];
    __syncthreads();

    int bt = blockIdx.x * 256 + tid;
    if (bt >= BT) return;
    const float* xs = x + (long)bt * N_IN;
    _Float16* fo = feat + (long)bt * 160;

    for (int j = 0; j < 5; ++j) {
        float xin[9];
#pragma unroll
        for (int q = 0; q < 9; ++q) {
            int ix = 8 * j - 1 + q;
            xin[q] = (ix >= 0 && ix < N_IN) ? xs[ix] : 0.f;
        }
        float h0[4][8];
#pragma unroll
        for (int qq = 0; qq < 4; ++qq)
#pragma unroll
            for (int o = 0; o < 8; ++o)
                h0[qq][o] = lrelu(s0w[o * 2] * xin[2 * qq] + s0w[o * 2 + 1] * xin[2 * qq + 1] + s0b[o]);
        float h1[2][16];
#pragma unroll
        for (int pp = 0; pp < 2; ++pp)
#pragma unroll
            for (int o = 0; o < 16; ++o) {
                float s = s1b[o];
#pragma unroll
                for (int i = 0; i < 8; ++i)
                    s += s1w[(o * 8 + i) * 2] * h0[2 * pp][i] + s1w[(o * 8 + i) * 2 + 1] * h0[2 * pp + 1][i];
                h1[pp][o] = lrelu(s);
            }
#pragma unroll
        for (int o = 0; o < 32; ++o) {
            float s = s2b[o];
#pragma unroll
            for (int i = 0; i < 16; ++i)
                s += s2w[(o * 16 + i) * 2] * h1[0][i] + s2w[(o * 16 + i) * 2 + 1] * h1[1][i];
            fo[o * 5 + j] = (_Float16)lrelu(s);   // feature index = o*5 + pos
        }
    }
}

// =====================================================================
// Weight f32->f16 conversion with K padding (pads to multiple of 32)
// =====================================================================
__global__ void wpad_f16(const float* __restrict__ w, _Float16* __restrict__ o,
                         int N, int Kin, int Kpad)
{
    int idx = blockIdx.x * 256 + threadIdx.x;
    if (idx >= N * Kpad) return;
    int n = idx / Kpad, k = idx % Kpad;
    o[idx] = (k < Kin) ? (_Float16)w[n * Kin + k] : (_Float16)0.f;
}

__global__ void wcat_f16(const float* __restrict__ wih, const float* __restrict__ whh,
                         _Float16* __restrict__ o)
{
    int idx = blockIdx.x * 256 + threadIdx.x;
    if (idx >= 1024 * 768) return;
    int n = idx / 768, k = idx % 768;
    o[idx] = (_Float16)(k < 512 ? wih[n * 512 + k] : whh[n * 256 + (k - 512)]);
}

__global__ void bcat_f32(const float* __restrict__ bih, const float* __restrict__ bhh,
                         float* __restrict__ o)
{
    int idx = blockIdx.x * 256 + threadIdx.x;
    if (idx < 1024) o[idx] = bih[idx] + bhh[idx];
}

// =====================================================================
// Tiled WMMA GEMM:  C[M,N] = act( A[M,K] * W[N,K]^T + bias )
// block = 256 threads (8 waves), 128x64 block tile, double-buffered LDS.
// wave grid 4(M) x 2(N); wave tile = 32x32 => 4 V_WMMA_F32_16X16X32_F16
// per K-step with a single workgroup barrier per step.
// K multiple of 32; M multiple of 128. N guarded by Wn.
// Writes optional f32 (strided, can target d_out directly) and/or f16.
// =====================================================================
__global__ __launch_bounds__(256) void gemm_wmma_f16(
    const _Float16* __restrict__ A, int lda,
    const _Float16* __restrict__ W, int Wn, int K,
    const float* __restrict__ bias,
    float* __restrict__ outF, int ldcF,
    _Float16* __restrict__ outH, int ldcH,
    int act)
{
    __shared__ __align__(16) _Float16 As[2][128 * 32];   // 16 KB
    __shared__ __align__(16) _Float16 Ws[2][64 * 32];    //  8 KB

    const int tid  = threadIdx.x;
    const int row0 = blockIdx.x * 128;
    const int n0   = blockIdx.y * 64;
    const int lane = tid & 31;
    const int wave = tid >> 5;
    const int wm   = wave >> 1;       // 0..3 : 32-row M sub-tile
    const int wn   = wave & 1;        // 0..1 : 32-col N sub-tile
    const int lrow = tid >> 2;        // 0..63 : LDS staging row
    const int lseg = tid & 3;         // 0..3  : 8-half segment

    v8f acc00 = {}, acc01 = {}, acc10 = {}, acc11 = {};

    const bool wvalid = (n0 + lrow) < Wn;

    // ---- stage tile 0 ----
    {
        const uint4* pa0 = (const uint4*)(A + (long)(row0 + lrow) * lda + lseg * 8);
        const uint4* pa1 = (const uint4*)(A + (long)(row0 + 64 + lrow) * lda + lseg * 8);
        uint4 av0 = *pa0;
        uint4 av1 = *pa1;
        uint4 wv = make_uint4(0u, 0u, 0u, 0u);
        if (wvalid) wv = *(const uint4*)(W + (long)(n0 + lrow) * K + lseg * 8);
        *(uint4*)&As[0][lrow * 32 + lseg * 8]        = av0;
        *(uint4*)&As[0][(64 + lrow) * 32 + lseg * 8] = av1;
        *(uint4*)&Ws[0][lrow * 32 + lseg * 8]        = wv;
    }
    __syncthreads();

    for (int k0 = 0; k0 < K; k0 += 32) {
        const int cur = (k0 >> 5) & 1;
        const int nxt = cur ^ 1;

        // ---- prefetch + stage next tile into the other buffer ----
        if (k0 + 32 < K) {
            const int k1 = k0 + 32;
            const uint4* pa0 = (const uint4*)(A + (long)(row0 + lrow) * lda + k1 + lseg * 8);
            const uint4* pa1 = (const uint4*)(A + (long)(row0 + 64 + lrow) * lda + k1 + lseg * 8);
            uint4 av0 = *pa0;
            uint4 av1 = *pa1;
            uint4 wv = make_uint4(0u, 0u, 0u, 0u);
            const uint4* pw = (const uint4*)(W + (long)(n0 + lrow) * K + k1 + lseg * 8);
            if (wvalid) wv = *pw;
            if (k1 + 32 < K) {                   // gfx1250 global_prefetch_b8 (near)
                __builtin_prefetch((const char*)pa0 + 64, 0, 3);
                __builtin_prefetch((const char*)pa1 + 64, 0, 3);
                __builtin_prefetch((const char*)pw  + 64, 0, 3);
            }
            *(uint4*)&As[nxt][lrow * 32 + lseg * 8]        = av0;
            *(uint4*)&As[nxt][(64 + lrow) * 32 + lseg * 8] = av1;
            *(uint4*)&Ws[nxt][lrow * 32 + lseg * 8]        = wv;
        }

        // ---- fragments from current buffer ----
        // A frag (16x32): lanes 0-15 => K {0..7,16..23}; lanes 16-31 => {8..15,24..31}
        v16h a0, a1;
        {
            int g = (lane >> 4) * 8;
            const _Float16* r0 = &As[cur][(wm * 32 + (lane & 15)) * 32];
            const _Float16* r1 = &As[cur][(wm * 32 + 16 + (lane & 15)) * 32];
            _Float16* a0p = (_Float16*)&a0;
            _Float16* a1p = (_Float16*)&a1;
#pragma unroll
            for (int i = 0; i < 8; ++i) { a0p[i] = r0[g + i];      a1p[i] = r1[g + i]; }
#pragma unroll
            for (int i = 0; i < 8; ++i) { a0p[8 + i] = r0[16 + g + i]; a1p[8 + i] = r1[16 + g + i]; }
        }
        // B frag (32x16): lane = N col; lanes 0-15 => K {0..7,16..23}; lanes 16-31 => {8..15,24..31}
        v16h b0, b1;
        {
            int g = (lane >> 4) * 8;
            const _Float16* r0 = &Ws[cur][(wn * 32 + (lane & 15)) * 32];
            const _Float16* r1 = &Ws[cur][(wn * 32 + 16 + (lane & 15)) * 32];
            _Float16* b0p = (_Float16*)&b0;
            _Float16* b1p = (_Float16*)&b1;
#pragma unroll
            for (int i = 0; i < 8; ++i) { b0p[i] = r0[g + i]; b1p[i] = r1[g + i]; b0p[8 + i] = r0[16 + g + i]; b1p[8 + i] = r1[16 + g + i]; }
        }

        acc00 = __builtin_amdgcn_wmma_f32_16x16x32_f16(false, a0, false, b0, (short)0, acc00, false, false);
        acc01 = __builtin_amdgcn_wmma_f32_16x16x32_f16(false, a0, false, b1, (short)0, acc01, false, false);
        acc10 = __builtin_amdgcn_wmma_f32_16x16x32_f16(false, a1, false, b0, (short)0, acc10, false, false);
        acc11 = __builtin_amdgcn_wmma_f32_16x16x32_f16(false, a1, false, b1, (short)0, acc11, false, false);

        __syncthreads();   // next-tile LDS stores complete; current reads done
    }

    // ---- epilogue: bias + LeakyReLU fused, strided f32 and/or f16 out ----
    const int nb = n0 + wn * 32 + (lane & 15);
#pragma unroll
    for (int am = 0; am < 2; ++am) {
        const int mrow = row0 + wm * 32 + am * 16 + (lane >> 4) * 8;
#pragma unroll
        for (int tgt = 0; tgt < 2; ++tgt) {
            int n = nb + tgt * 16;
            if (n >= Wn) continue;
            float bi = bias ? bias[n] : 0.f;
            v8f acc = am ? (tgt ? acc11 : acc10) : (tgt ? acc01 : acc00);
#pragma unroll
            for (int r = 0; r < 8; ++r) {
                float v = acc[r] + bi;
                if (act) v = lrelu(v);
                long m = mrow + r;
                if (outF) outF[m * (long)ldcF + n] = v;
                if (outH) outH[m * (long)ldcH + n] = (_Float16)v;
            }
        }
    }
}

// =====================================================================
// Elementwise / glue kernels
// =====================================================================
__global__ void concat2_f16(const _Float16* __restrict__ s1, int ld1, int d1,
                            const _Float16* __restrict__ s2, int ld2, int d2,
                            _Float16* __restrict__ o, int ldo, int n)
{
    int idx = blockIdx.x * 256 + threadIdx.x;
    if (idx >= n) return;
    int w = d1 + d2;
    int b = idx / w, j = idx % w;
    o[(long)b * ldo + j] = (j < d1) ? s1[(long)b * ld1 + j] : s2[(long)b * ld2 + (j - d1)];
}

__global__ void storeh_kernel(const float* __restrict__ h, float* __restrict__ outH, int ldo)
{
    int idx = blockIdx.x * 256 + threadIdx.x;
    if (idx >= B_ * HID) return;
    int b = idx >> 8, j = idx & 255;
    outH[(long)b * ldo + j] = h[idx];
}

// softmax(logits) -> prob ; soft gumbel-softmax -> cate ; one row per thread
__global__ void gumbel_kernel(const float* __restrict__ logits, const float* __restrict__ u,
                              const float* __restrict__ temp,
                              float* __restrict__ out_log, float* __restrict__ out_prob,
                              float* __restrict__ out_cate, _Float16* __restrict__ cate16,
                              int ldo)
{
    int b = blockIdx.x * 128 + threadIdx.x;
    if (b >= B_) return;
    float l[CD], y[CD];
    float mx = -1e30f;
#pragma unroll
    for (int i = 0; i < CD; ++i) { l[i] = logits[b * CD + i]; mx = fmaxf(mx, l[i]); }
    float s = 0.f;
#pragma unroll
    for (int i = 0; i < CD; ++i) s += __expf(l[i] - mx);
    float inv = 1.f / s;
#pragma unroll
    for (int i = 0; i < CD; ++i) {
        out_prob[(long)b * ldo + i] = __expf(l[i] - mx) * inv;
        out_log[(long)b * ldo + i] = l[i];
    }
    float tpr = *temp;
    float mx2 = -1e30f;
#pragma unroll
    for (int i = 0; i < CD; ++i) {
        float g = -__logf(-__logf(u[(long)b * CD + i] + 1e-20f) + 1e-20f);
        y[i] = (l[i] + g) / tpr;
        mx2 = fmaxf(mx2, y[i]);
    }
    s = 0.f;
#pragma unroll
    for (int i = 0; i < CD; ++i) s += __expf(y[i] - mx2);
    inv = 1.f / s;
#pragma unroll
    for (int i = 0; i < CD; ++i) {
        float cv = __expf(y[i] - mx2) * inv;
        out_cate[(long)b * ldo + i] = cv;
        cate16[b * CD + i] = (_Float16)cv;
    }
}

// z = mean + eps * exp(0.5*logvar); reads mean/logvar from strided d_out zones
__global__ void z_kernel(const float* __restrict__ meanO, const float* __restrict__ lvO,
                         const float* __restrict__ eps, float* __restrict__ zO,
                         _Float16* __restrict__ z16, int ldo)
{
    int idx = blockIdx.x * 256 + threadIdx.x;
    if (idx >= B_ * ZD) return;
    int b = idx >> 6, d = idx & 63;
    float m  = meanO[(long)b * ldo + d];
    float lv = lvO[(long)b * ldo + d];
    float z  = m + eps[idx] * __expf(0.5f * lv);
    zO[(long)b * ldo + d] = z;
    z16[idx] = (_Float16)z;
}

// torch LSTMCell gate order: i, f, g, o
__global__ void lstm_kernel(const float* __restrict__ gates, float* __restrict__ h,
                            float* __restrict__ c, _Float16* __restrict__ h16)
{
    int idx = blockIdx.x * 256 + threadIdx.x;
    if (idx >= B_ * HID) return;
    int b = idx >> 8, j = idx & 255;
    const float* gr = gates + (long)b * 1024;
    float ig = gr[j], fg = gr[256 + j], gg = gr[512 + j], og = gr[768 + j];
    float cn = sigm(fg) * c[idx] + sigm(ig) * tanhf(gg);
    float hn = sigm(og) * tanhf(cn);
    c[idx] = cn;
    h[idx] = hn;
    h16[idx] = (_Float16)hn;
}

// =====================================================================
// Host launcher
// =====================================================================
extern "C" void kernel_launch(void* const* d_in, const int* in_sizes, int n_in,
                              void* d_out, int out_size, void* d_ws, size_t ws_size,
                              hipStream_t stream)
{
    (void)in_sizes; (void)n_in; (void)out_size; (void)ws_size;
    const float* x    = (const float*)d_in[0];
    const float* temp = (const float*)d_in[1];
    const float* gum  = (const float*)d_in[2];
    const float* geps = (const float*)d_in[3];
    const float* P[28];
    for (int i = 0; i < 28; ++i) P[i] = (const float*)d_in[4 + i];
    enum { C0W, C0B, C1W, C1B, C2W, C2B, FC0W, FC0B, FC1W, FC1B,
           XHC0W, XHC0B, XHC1W, XHC1B, PHIW, PHIB, WIH, BIH, WHH, BHH,
           M0W, M0B, M1W, M1B, V0W, V0B, V1W, V1B };

    char* ws = (char*)d_ws;
    size_t off = 0;
    auto alloc = [&](size_t bytes) -> char* {
        char* p = ws + off;
        off = (off + bytes + 255) & ~(size_t)255;
        return p;
    };

    // -------- persistent region --------
    _Float16* xhat   = (_Float16*)alloc((size_t)BT * ENC * 2);   // f16 encoder output, whole run
    _Float16* w_fc0  = (_Float16*)alloc(256 * 160 * 2);
    _Float16* w_fc1  = (_Float16*)alloc(128 * 256 * 2);
    _Float16* w_xhc0 = (_Float16*)alloc(256 * 384 * 2);
    _Float16* w_xhc1 = (_Float16*)alloc(16 * 256 * 2);
    _Float16* w_m0   = (_Float16*)alloc(256 * 416 * 2);          // K padded 400->416
    _Float16* w_m1   = (_Float16*)alloc(64 * 256 * 2);
    _Float16* w_v0   = (_Float16*)alloc(256 * 416 * 2);
    _Float16* w_v1   = (_Float16*)alloc(64 * 256 * 2);
    _Float16* w_phi  = (_Float16*)alloc(512 * 192 * 2);
    _Float16* w_cat  = (_Float16*)alloc(1024 * 768 * 2);         // [wih | whh]
    float*    b_cat  = (float*)alloc(1024 * 4);

    // -------- time-overlapped union region --------
    size_t ubase = off;
    // encoder scratch (dead after the fc1 GEMM)
    _Float16* convfeat = (_Float16*)(ws + ubase);                          // BT*160*2 = 0x1900000
    _Float16* fc0out   = (_Float16*)(ws + ubase + (size_t)0x1900000);      // BT*256*2 = 0x2800000
    // RNN scratch (reuses the same region, used only after encoder drains)
    size_t r = ubase;
    auto ralloc = [&](size_t bytes) -> char* {
        char* p = ws + r;
        r = (r + bytes + 255) & ~(size_t)255;
        return p;
    };
    float*    hF      = (float*)ralloc((size_t)B_ * HID * 4);
    float*    cF      = (float*)ralloc((size_t)B_ * HID * 4);
    _Float16* h16     = (_Float16*)ralloc((size_t)B_ * HID * 2);
    _Float16* xht     = (_Float16*)ralloc((size_t)B_ * 384 * 2);
    _Float16* hid     = (_Float16*)ralloc((size_t)B_ * 256 * 2);
    float*    logitsB = (float*)ralloc((size_t)B_ * CD * 4);
    _Float16* cate16  = (_Float16*)ralloc((size_t)B_ * CD * 2);
    _Float16* xhc     = (_Float16*)ralloc((size_t)B_ * 416 * 2);
    _Float16* mh      = (_Float16*)ralloc((size_t)B_ * 256 * 2);
    _Float16* vh      = (_Float16*)ralloc((size_t)B_ * 256 * 2);
    _Float16* z16     = (_Float16*)ralloc((size_t)B_ * ZD * 2);
    _Float16* xz      = (_Float16*)ralloc((size_t)B_ * 192 * 2);
    _Float16* phiB    = (_Float16*)ralloc((size_t)B_ * 512 * 2);
    _Float16* ph      = (_Float16*)ralloc((size_t)B_ * 768 * 2);
    float*    gates   = (float*)ralloc((size_t)B_ * 1024 * 4);

    // -------- output zones (z, mean, logvar, h, cate, logits, prob), each [B,T,D] --------
    float* outF  = (float*)d_out;
    float* zO    = outF;
    float* meanO = outF + (size_t)BT * 64;
    float* lvO   = outF + (size_t)BT * 128;
    float* hO    = outF + (size_t)BT * 192;
    float* cateO = outF + (size_t)BT * 448;
    float* logO  = outF + (size_t)BT * 464;
    float* probO = outF + (size_t)BT * 480;

    // -------- weight conversion (f32 -> f16, K-padded) --------
    auto wpad = [&](const float* w, _Float16* o, int N, int Kin, int Kpad) {
        int n = N * Kpad;
        wpad_f16<<<(n + 255) / 256, 256, 0, stream>>>(w, o, N, Kin, Kpad);
    };
    wpad(P[FC0W],  w_fc0,  256, 160, 160);
    wpad(P[FC1W],  w_fc1,  128, 256, 256);
    wpad(P[XHC0W], w_xhc0, 256, 384, 384);
    wpad(P[XHC1W], w_xhc1, 16,  256, 256);
    wpad(P[M0W],   w_m0,   256, 400, 416);
    wpad(P[M1W],   w_m1,   64,  256, 256);
    wpad(P[V0W],   w_v0,   256, 400, 416);
    wpad(P[V1W],   w_v1,   64,  256, 256);
    wpad(P[PHIW],  w_phi,  512, 192, 192);
    wcat_f16<<<(1024 * 768 + 255) / 256, 256, 0, stream>>>(P[WIH], P[WHH], w_cat);
    bcat_f32<<<4, 256, 0, stream>>>(P[BIH], P[BHH], b_cat);

    // -------- encoder: conv -> fc0(160->256) -> fc1(256->128), all leaky --------
    conv_enc_kernel<<<BT / 256, 256, 0, stream>>>(x, P[C0W], P[C0B], P[C1W], P[C1B],
                                                  P[C2W], P[C2B], convfeat);
    gemm_wmma_f16<<<dim3(BT / 128, 4), 256, 0, stream>>>(convfeat, 160, w_fc0, 256, 160,
        P[FC0B], nullptr, 0, fc0out, 256, 1);
    gemm_wmma_f16<<<dim3(BT / 128, 2), 256, 0, stream>>>(fc0out, 256, w_fc1, 128, 256,
        P[FC1B], nullptr, 0, xhat, 128, 1);

    // -------- RNN state init (after encoder: region is reused) --------
    (void)hipMemsetAsync(hF,  0, (size_t)B_ * HID * 4, stream);
    (void)hipMemsetAsync(cF,  0, (size_t)B_ * HID * 4, stream);
    (void)hipMemsetAsync(h16, 0, (size_t)B_ * HID * 2, stream);
    (void)hipMemsetAsync(xhc, 0, (size_t)B_ * 416 * 2, stream);   // zero K-pad lanes

    // -------- recurrence --------
    for (int t = 0; t < T_; ++t) {
        const _Float16* x_t = xhat + (size_t)t * ENC;   // row stride T_*ENC

        // h output is the PRE-update h (reference stores carry h)
        storeh_kernel<<<B_ * HID / 256, 256, 0, stream>>>(hF, hO + (size_t)t * HID, T_ * HID);

        // xht = [x_t | h]
        concat2_f16<<<B_ * 384 / 256, 256, 0, stream>>>(x_t, T_ * ENC, ENC, h16, HID, HID,
                                                        xht, 384, B_ * 384);
        // hid = leaky(xht @ xhc0^T + b)
        gemm_wmma_f16<<<dim3(B_ / 128, 4), 256, 0, stream>>>(xht, 384, w_xhc0, 256, 384,
            P[XHC0B], nullptr, 0, hid, 256, 1);
        // logits = leaky(hid @ xhc1^T + b)  (f32 for softmax)
        gemm_wmma_f16<<<dim3(B_ / 128, 1), 256, 0, stream>>>(hid, 256, w_xhc1, CD, 256,
            P[XHC1B], logitsB, CD, nullptr, 0, 1);
        // prob / gumbel-softmax cate / logits out
        gumbel_kernel<<<(B_ + 127) / 128, 128, 0, stream>>>(logitsB,
            gum + (size_t)t * B_ * CD, temp,
            logO + (size_t)t * CD, probO + (size_t)t * CD, cateO + (size_t)t * CD,
            cate16, T_ * CD);
        // xhc = [xht | cate] (padded 400->416, pad pre-zeroed)
        concat2_f16<<<B_ * 400 / 256, 256, 0, stream>>>(xht, 384, 384, cate16, CD, CD,
                                                        xhc, 416, B_ * 400);
        // mean = leaky(leaky(xhc@m0)@m1) -> straight into d_out (strided)
        gemm_wmma_f16<<<dim3(B_ / 128, 4), 256, 0, stream>>>(xhc, 416, w_m0, 256, 416,
            P[M0B], nullptr, 0, mh, 256, 1);
        gemm_wmma_f16<<<dim3(B_ / 128, 1), 256, 0, stream>>>(mh, 256, w_m1, ZD, 256,
            P[M1B], meanO + (size_t)t * ZD, T_ * ZD, nullptr, 0, 1);
        // logvar path
        gemm_wmma_f16<<<dim3(B_ / 128, 4), 256, 0, stream>>>(xhc, 416, w_v0, 256, 416,
            P[V0B], nullptr, 0, vh, 256, 1);
        gemm_wmma_f16<<<dim3(B_ / 128, 1), 256, 0, stream>>>(vh, 256, w_v1, ZD, 256,
            P[V1B], lvO + (size_t)t * ZD, T_ * ZD, nullptr, 0, 1);
        // z = mean + eps*exp(0.5*logvar)
        z_kernel<<<B_ * ZD / 256, 256, 0, stream>>>(meanO + (size_t)t * ZD,
            lvO + (size_t)t * ZD, geps + (size_t)t * B_ * ZD,
            zO + (size_t)t * ZD, z16, T_ * ZD);
        // xz = [x_t | z]
        concat2_f16<<<B_ * 192 / 256, 256, 0, stream>>>(x_t, T_ * ENC, ENC, z16, ZD, ZD,
                                                        xz, 192, B_ * 192);
        // phi = leaky(xz @ phi^T + b)
        gemm_wmma_f16<<<dim3(B_ / 128, 8), 256, 0, stream>>>(xz, 192, w_phi, 512, 192,
            P[PHIB], nullptr, 0, phiB, 512, 1);
        // gates = [phi | h] @ [wih | whh]^T + (bih + bhh)   (single K=768 GEMM)
        concat2_f16<<<B_ * 768 / 256, 256, 0, stream>>>(phiB, 512, 512, h16, HID, HID,
                                                        ph, 768, B_ * 768);
        gemm_wmma_f16<<<dim3(B_ / 128, 16), 256, 0, stream>>>(ph, 768, w_cat, 1024, 768,
            b_cat, gates, 1024, nullptr, 0, 0);
        // LSTM cell update
        lstm_kernel<<<B_ * HID / 256, 256, 0, stream>>>(gates, hF, cF, h16);
    }
}
